// MultiHeadAttentionLayer_40956808134797
// MI455X (gfx1250) — matrix operand order repaired and checked by reading, not compiled, not run on hardware
//
#include <hip/hip_runtime.h>
#include <hip/hip_bf16.h>

// ---------------------------------------------------------------------------
// MHA layer for MI455X (gfx1250): LN -> QKV proj -> flash attention -> out proj
// - All GEMM math on V_WMMA_F32_16X16X32_BF16 (wave32, f32 accum).
// - GEMM A tiles streamed via GLOBAL_LOAD_ASYNC_TO_LDS_B128 (ASYNCcnt).
// - Attention K tiles streamed via the Tensor Data Mover (TENSOR_LOAD_TO_LDS,
//   TENSORcnt), one descriptor per workgroup per tile.
// - Double-buffered LDS pipelines: DMA tile i+1 while WMMAs consume tile i.
// - All LDS fragment reads are ds_load_b128 (transposed B layouts in LDS).
// ---------------------------------------------------------------------------

typedef __attribute__((ext_vector_type(16))) __bf16 v16bf;
typedef __attribute__((ext_vector_type(8)))  float  v8f;
typedef __attribute__((ext_vector_type(4)))  unsigned u32x4;
typedef __attribute__((ext_vector_type(8)))  unsigned u32x8;

#define HID 512
#define SEQ 2048
#define BAT 2
#define NH  8
#define HD  64
#define MROWS (BAT * SEQ)   // 4096

#define TM 128
#define TN 64
#define TK 64               // K-step per LDS stage (2 WMMA k-substeps)
#define NIT (HID / TK)      // 8 pipeline stages

union Pack8  { __bf16 h[8];  uint4 q;    };
union Pack16 { __bf16 h[16]; uint4 q[2]; };
union Frag16 { uint4 q[2];   v16bf v;    };

static __device__ inline v8f zero8() {
  v8f z;
#pragma unroll
  for (int i = 0; i < 8; ++i) z[i] = 0.0f;
  return z;
}

static __device__ inline v8f wmma_bf16(v16bf a, v16bf b, v8f c) {
  // D = A(16x32 bf16) x B(32x16 bf16) + C(16x16 f32)
  return __builtin_amdgcn_wmma_f32_16x16x32_bf16(false, a, false, b,
                                                 (short)0, c, false, false);
}

// ---- async global->LDS (ASYNCcnt) -----------------------------------------
static __device__ inline void async_copy_b128(const void* gptr, void* lptr) {
  const unsigned lds = (unsigned)(size_t)lptr;  // low 32 bits of generic LDS ptr
  asm volatile("global_load_async_to_lds_b128 %0, %1, off"
               :: "v"(lds), "v"(gptr) : "memory");
}
static __device__ inline void async_wait_all() {
  asm volatile("s_wait_asynccnt 0" ::: "memory");
}

// ---- Tensor Data Mover: 2D tile global->LDS (TENSORcnt) --------------------
// D# groups per CDNA5 ISA §8.3/8.4 (user mode, no gather/iterate/pad,
// data_size = 2 bytes, 2D tensor -> groups 2/3 NULL).
static __device__ inline void tdm_load_tile_2d(unsigned lds_byte_addr,
                                               const void* gaddr,
                                               unsigned tensorW, unsigned tensorH,
                                               unsigned tileW, unsigned tileH,
                                               unsigned strideElems) {
  const unsigned long long ga = (unsigned long long)(size_t)gaddr;
  u32x4 g0;
  g0[0] = 1u;                                    // count=1, user mode
  g0[1] = lds_byte_addr;                         // lds_addr
  g0[2] = (unsigned)ga;                          // global_addr[31:0]
  g0[3] = (unsigned)((ga >> 32) & 0x01ffffffu)   // global_addr[56:32]
        | (2u << 30);                            // type = 2 ("image")
  u32x8 g1;
  g1[0] = 1u << 16;                              // data_size = 1 -> 2 bytes
  g1[1] = (tensorW & 0xffffu) << 16;             // tensor_dim0[15:0] @ bits 63:48
  g1[2] = (tensorW >> 16) | ((tensorH & 0xffffu) << 16);  // dim0 hi | dim1 lo
  g1[3] = (tensorH >> 16) | ((tileW & 0xffffu) << 16);    // dim1 hi | tile_dim0
  g1[4] = tileH & 0xffffu;                       // tile_dim1 (tile_dim2 = 0)
  g1[5] = strideElems;                           // tensor_dim0_stride[31:0]
  g1[6] = 0;                                     // stride hi / dim1_stride lo
  g1[7] = 0;                                     // dim1_stride hi
  asm volatile("tensor_load_to_lds %0, %1" :: "s"(g0), "s"(g1) : "memory");
}
static __device__ inline void tdm_wait_all() {
  __builtin_amdgcn_s_wait_tensorcnt(0);
}

// A fragment 16x32 (row-major, leading dim ld):
// lane&15 = row; lanes 0-15 K = k0+{0..7,16..23}, lanes 16-31 K = k0+{8..15,24..31}
static __device__ inline v16bf load_a16(const __bf16* __restrict__ p, int ld,
                                        int row0, int k0, int lane) {
  const int r  = row0 + (lane & 15);
  const int kb = k0 + ((lane >> 4) & 1) * 8;
  const __bf16* s = p + (size_t)r * ld + kb;
  Frag16 u;
  u.q[0] = *(const uint4*)s;
  u.q[1] = *(const uint4*)(s + 16);
  return u.v;
}

// B fragment 32x16 from a TRANSPOSED ([N][K] row-major) source: contiguous.
static __device__ inline v16bf load_bT16(const __bf16* __restrict__ p, int ld,
                                         int k0, int col0, int lane) {
  const int n  = col0 + (lane & 15);
  const int kb = k0 + ((lane >> 4) & 1) * 8;
  const __bf16* s = p + (size_t)n * ld + kb;
  Frag16 u;
  u.q[0] = *(const uint4*)s;
  u.q[1] = *(const uint4*)(s + 16);
  return u.v;
}

// ---------------------------------------------------------------------------
// Kernel 1: LayerNorm(q|k|v) -> bf16.
// ---------------------------------------------------------------------------
__global__ __launch_bounds__(256) void mha_ln_bf16(
    const float* __restrict__ q, const float* __restrict__ k,
    const float* __restrict__ v, const float* __restrict__ g,
    const float* __restrict__ bb, __bf16* __restrict__ oq,
    __bf16* __restrict__ ok, __bf16* __restrict__ ov) {
  const int row   = blockIdx.x;
  const int which = blockIdx.y;
  const float* x = (which == 0) ? q : (which == 1) ? k : v;
  __bf16*      o = (which == 0) ? oq : (which == 1) ? ok : ov;
  x += (size_t)row * HID;
  o += (size_t)row * HID;

  __shared__ float red[256];
  const int t = threadIdx.x;
  const float a0 = x[t], a1 = x[t + 256];

  red[t] = a0 + a1;
  __syncthreads();
#pragma unroll
  for (int off = 128; off > 0; off >>= 1) {
    if (t < off) red[t] += red[t + off];
    __syncthreads();
  }
  const float mu = red[0] * (1.0f / HID);
  __syncthreads();

  const float d0 = a0 - mu, d1 = a1 - mu;
  red[t] = d0 * d0 + d1 * d1;
  __syncthreads();
#pragma unroll
  for (int off = 128; off > 0; off >>= 1) {
    if (t < off) red[t] += red[t + off];
    __syncthreads();
  }
  const float rstd = __builtin_amdgcn_rsqf(red[0] * (1.0f / HID) + 1e-5f);

  o[t]       = (__bf16)(d0 * rstd * g[t]       + bb[t]);
  o[t + 256] = (__bf16)(d1 * rstd * g[t + 256] + bb[t + 256]);
}

// ---------------------------------------------------------------------------
// Kernel 2/4: C = A(bf16 [4096x512]) * W(f32 [512x512]) + bias.
// Double-buffered: async-DMA A tile i+1 + issue W loads while 8 WMMAs chew
// tile i.  W stored transposed in LDS so B frags are contiguous b128 reads.
// ---------------------------------------------------------------------------
__global__ __launch_bounds__(256) void mha_gemm512(
    const __bf16* __restrict__ A, const float* __restrict__ W,
    const float* __restrict__ bias, __bf16* __restrict__ outH,
    float* __restrict__ outF) {
  __shared__ __align__(16) __bf16 As[2 * TM * TK];   // 2 x 16 KB
  __shared__ __align__(16) __bf16 BsT[2 * TN * TK];  // 2 x  8 KB

  const int t    = threadIdx.x;
  const int lane = t & 31;
  const int wave = t >> 5;
  const int wm   = wave & 3;   // 4 waves along M (32 rows each)
  const int wn   = wave >> 2;  // 2 waves along N (32 cols each)
  const int mBase = blockIdx.y * TM;
  const int nBase = blockIdx.x * TN;

  const int ar = t >> 1, ac = (t & 1) * 32;   // A: 32 bf16 per thread
  const int bn = t & 63, bk = (t >> 6) * 16;  // W: one column, 16 k's

  v8f acc[2][2];
#pragma unroll
  for (int mi = 0; mi < 2; ++mi)
#pragma unroll
    for (int ni = 0; ni < 2; ++ni) acc[mi][ni] = zero8();

  // ---- prologue: stage tile 0 into buffer 0 ----
  {
    const __bf16* src = A + (size_t)(mBase + ar) * HID + ac;
    __bf16* dst = As + ar * TK + ac;
#pragma unroll
    for (int j = 0; j < 4; ++j) async_copy_b128(src + j * 8, dst + j * 8);

    const float* wsrc = W + (size_t)bk * HID + nBase + bn;
    Pack16 pk;
#pragma unroll
    for (int j = 0; j < 16; ++j) pk.h[j] = (__bf16)wsrc[(size_t)j * HID];
    uint4* bdst = (uint4*)(BsT + bn * TK + bk);
    bdst[0] = pk.q[0];
    bdst[1] = pk.q[1];
  }
  async_wait_all();
  __syncthreads();

  // ---- pipelined main loop ----
  for (int i = 0; i < NIT; ++i) {
    const __bf16* Ac = As  + (i & 1) * (TM * TK);
    const __bf16* Bc = BsT + (i & 1) * (TN * TK);
    __bf16* An = As  + ((i + 1) & 1) * (TM * TK);
    __bf16* Bn = BsT + ((i + 1) & 1) * (TN * TK);
    const bool more = (i + 1 < NIT);

    float wreg[16];
    if (more) {
      const int kk = (i + 1) * TK;
      const __bf16* src = A + (size_t)(mBase + ar) * HID + kk + ac;
      __bf16* dst = An + ar * TK + ac;
#pragma unroll
      for (int j = 0; j < 4; ++j) async_copy_b128(src + j * 8, dst + j * 8);
      const float* wsrc = W + (size_t)(kk + bk) * HID + nBase + bn;
#pragma unroll
      for (int j = 0; j < 16; ++j) wreg[j] = wsrc[(size_t)j * HID];
    }

    v16bf af[2][2], bfr[2][2];
#pragma unroll
    for (int mi = 0; mi < 2; ++mi)
#pragma unroll
      for (int ks = 0; ks < 2; ++ks)
        af[mi][ks] = load_a16(Ac, TK, wm * 32 + mi * 16, ks * 32, lane);
#pragma unroll
    for (int ni = 0; ni < 2; ++ni)
#pragma unroll
      for (int ks = 0; ks < 2; ++ks)
        bfr[ni][ks] = load_bT16(Bc, TK, ks * 32, wn * 32 + ni * 16, lane);
#pragma unroll
    for (int mi = 0; mi < 2; ++mi)
#pragma unroll
      for (int ni = 0; ni < 2; ++ni)
#pragma unroll
        for (int ks = 0; ks < 2; ++ks)
          acc[mi][ni] = wmma_bf16(af[mi][ks], bfr[ni][ks], acc[mi][ni]);

    if (more) {
      Pack16 pk;
#pragma unroll
      for (int j = 0; j < 16; ++j) pk.h[j] = (__bf16)wreg[j];
      uint4* bdst = (uint4*)(Bn + bn * TK + bk);
      bdst[0] = pk.q[0];
      bdst[1] = pk.q[1];
    }
    async_wait_all();
    __syncthreads();
  }

  // ---- epilogue: bias + store ----
#pragma unroll
  for (int mi = 0; mi < 2; ++mi) {
#pragma unroll
    for (int ni = 0; ni < 2; ++ni) {
      const int n  = nBase + wn * 32 + ni * 16 + (lane & 15);
      const float bv = bias[n];
      const int m0 = mBase + wm * 32 + mi * 16 + ((lane >> 4) & 1) * 8;
#pragma unroll
      for (int r = 0; r < 8; ++r) {
        const float val = acc[mi][ni][r] + bv;
        const int m = m0 + r;
        if (outF) {
          outF[(size_t)m * HID + n] = val;
        } else {
          const int b = m >> 11, s = m & (SEQ - 1);
          const int h = n >> 6, d = n & (HD - 1);
          outH[(((size_t)(b * NH + h)) * SEQ + s) * HD + d] = (__bf16)val;
        }
      }
    }
  }
}

// ---------------------------------------------------------------------------
// Kernel 3: flash attention per (b,h).  8 waves, each owns a 16x64 Q tile.
// K tiles DMA'd by the Tensor Data Mover (one descriptor / tile / workgroup),
// V tiles transposed on load; both double-buffered against the WMMA phase.
// ---------------------------------------------------------------------------
__global__ __launch_bounds__(256) void mha_flash_attn(
    const __bf16* __restrict__ Qh, const __bf16* __restrict__ Kh,
    const __bf16* __restrict__ Vh, __bf16* __restrict__ AttnO) {
  __shared__ __align__(16) __bf16 Ks[2 * 64 * HD];   // 2 x 8 KB, [kv][d]
  __shared__ __align__(16) __bf16 VsT[2 * HD * 64];  // 2 x 8 KB, [d][kv]
  __shared__ __align__(16) __bf16 Ps[8 * 16 * HD];   // 16 KB per-wave P scratch

  const int t    = threadIdx.x;
  const int lane = t & 31;
  const int wave = t >> 5;
  const int bh   = blockIdx.y;                 // b*8 + h
  const size_t base = (size_t)bh * SEQ * HD;
  const int qr = blockIdx.x * 128 + wave * 16;

  const int vd = t & 63, vk = (t >> 6) * 16;   // V: column vd, 16 kv's

  v16bf aq[2];
#pragma unroll
  for (int ks = 0; ks < 2; ++ks) aq[ks] = load_a16(Qh + base, HD, qr, ks * 32, lane);

  const float SC = 0.125f * 1.44269504088896340736f;  // 1/sqrt(64) * log2(e)

  float mrow[8], lrow[8];
#pragma unroll
  for (int r = 0; r < 8; ++r) { mrow[r] = -1e30f; lrow[r] = 0.0f; }
  v8f accO[4];
#pragma unroll
  for (int nt = 0; nt < 4; ++nt) accO[nt] = zero8();

  __bf16* myP = Ps + wave * 16 * HD;
  const int col   = lane & 15;
  const int rowhi = ((lane >> 4) & 1) * 8;
  const int NT = SEQ / 64;

  // ---- prologue: stage tile 0 ----
  if (wave == 0)
    tdm_load_tile_2d((unsigned)(size_t)Ks, Kh + base, HD, SEQ, HD, 64, HD);
  {
    const __bf16* src = Vh + base + (size_t)vk * HD + vd;
    Pack16 pk;
#pragma unroll
    for (int j = 0; j < 16; ++j) pk.h[j] = src[(size_t)j * HD];
    uint4* dst = (uint4*)(VsT + vd * 64 + vk);
    dst[0] = pk.q[0];
    dst[1] = pk.q[1];
  }
  if (wave == 0) tdm_wait_all();
  __syncthreads();

  for (int kt = 0; kt < NT; ++kt) {
    const __bf16* Kc = Ks  + (kt & 1) * (64 * HD);
    const __bf16* Vc = VsT + (kt & 1) * (HD * 64);
    __bf16* Knx = Ks  + ((kt + 1) & 1) * (64 * HD);
    __bf16* Vnx = VsT + ((kt + 1) & 1) * (HD * 64);
    const bool more = (kt + 1 < NT);

    Pack16 vreg;
    if (more) {
      if (wave == 0)
        tdm_load_tile_2d((unsigned)(size_t)Knx,
                         Kh + base + (size_t)(kt + 1) * 64 * HD,
                         HD, SEQ, HD, 64, HD);
      const __bf16* src = Vh + base + (size_t)((kt + 1) * 64 + vk) * HD + vd;
#pragma unroll
      for (int j = 0; j < 16; ++j) vreg.h[j] = src[(size_t)j * HD];
    }

    // scores S = Q (16x64) @ K^T : 4 n-tiles x 2 k-steps
    v8f sc[4];
#pragma unroll
    for (int nt = 0; nt < 4; ++nt) {
      sc[nt] = zero8();
#pragma unroll
      for (int ks = 0; ks < 2; ++ks)
        sc[nt] = wmma_bf16(aq[ks], load_bT16(Kc, HD, ks * 32, nt * 16, lane), sc[nt]);
    }

    // scale + running row max
    float mx[8];
#pragma unroll
    for (int r = 0; r < 8; ++r) mx[r] = -1e30f;
#pragma unroll
    for (int nt = 0; nt < 4; ++nt)
#pragma unroll
      for (int r = 0; r < 8; ++r) {
        const float s = sc[nt][r] * SC;
        sc[nt][r] = s;
        mx[r] = fmaxf(mx[r], s);
      }
#pragma unroll
    for (int r = 0; r < 8; ++r)
#pragma unroll
      for (int off = 1; off < 16; off <<= 1)
        mx[r] = fmaxf(mx[r], __shfl_xor(mx[r], off, 32));

    float mnew[8], alpha[8], rs[8];
#pragma unroll
    for (int r = 0; r < 8; ++r) {
      mnew[r]  = fmaxf(mrow[r], mx[r]);
      alpha[r] = __builtin_amdgcn_exp2f(mrow[r] - mnew[r]);
      mrow[r]  = mnew[r];
      rs[r]    = 0.0f;
    }

    // P = 2^(s - m); row sums; stage P bf16 in per-wave LDS (A-layout source)
#pragma unroll
    for (int nt = 0; nt < 4; ++nt)
#pragma unroll
      for (int r = 0; r < 8; ++r) {
        const float p = __builtin_amdgcn_exp2f(sc[nt][r] - mnew[r]);
        rs[r] += p;
        myP[(rowhi + r) * HD + nt * 16 + col] = (__bf16)p;
      }
#pragma unroll
    for (int r = 0; r < 8; ++r) {
#pragma unroll
      for (int off = 1; off < 16; off <<= 1) rs[r] += __shfl_xor(rs[r], off, 32);
      lrow[r] = lrow[r] * alpha[r] + rs[r];
    }
#pragma unroll
    for (int nt = 0; nt < 4; ++nt)
#pragma unroll
      for (int r = 0; r < 8; ++r) accO[nt][r] *= alpha[r];

    // O += P (16x64) @ V (64x64)
    v16bf ap[2];
#pragma unroll
    for (int ks = 0; ks < 2; ++ks) ap[ks] = load_a16(myP, HD, 0, ks * 32, lane);
#pragma unroll
    for (int nt = 0; nt < 4; ++nt)
#pragma unroll
      for (int ks = 0; ks < 2; ++ks)
        accO[nt] = wmma_bf16(ap[ks], load_bT16(Vc, 64, ks * 32, nt * 16, lane), accO[nt]);

    if (more) {
      uint4* dst = (uint4*)(Vnx + vd * 64 + vk);
      dst[0] = vreg.q[0];
      dst[1] = vreg.q[1];
    }
    if (wave == 0) tdm_wait_all();
    __syncthreads();
  }

  // normalize by row sum and store bf16 row-major [4096 x 512]
  const int b = bh >> 3, h = bh & 7;
  float inv[8];
#pragma unroll
  for (int r = 0; r < 8; ++r) inv[r] = 1.0f / lrow[r];
#pragma unroll
  for (int nt = 0; nt < 4; ++nt) {
    const int n = h * HD + nt * 16 + col;
#pragma unroll
    for (int r = 0; r < 8; ++r) {
      const int m = b * SEQ + qr + rowhi + r;
      AttnO[(size_t)m * HID + n] = (__bf16)(accO[nt][r] * inv[r]);
    }
  }
}

// ---------------------------------------------------------------------------
extern "C" void kernel_launch(void* const* d_in, const int* in_sizes, int n_in,
                              void* d_out, int out_size, void* d_ws, size_t ws_size,
                              hipStream_t stream) {
  (void)in_sizes; (void)n_in; (void)out_size; (void)ws_size;
  const float* q   = (const float*)d_in[0];
  const float* k   = (const float*)d_in[1];
  const float* v   = (const float*)d_in[2];
  const float* lng = (const float*)d_in[3];
  const float* lnb = (const float*)d_in[4];
  const float* Wq  = (const float*)d_in[5];
  const float* bq  = (const float*)d_in[6];
  const float* Wk  = (const float*)d_in[7];
  const float* bk  = (const float*)d_in[8];
  const float* Wv  = (const float*)d_in[9];
  const float* bv  = (const float*)d_in[10];
  const float* Wo  = (const float*)d_in[11];
  const float* bo  = (const float*)d_in[12];

  char* ws = (char*)d_ws;
  const size_t SZ = (size_t)MROWS * HID * sizeof(__bf16);  // 4 MiB per buffer
  __bf16* LNq = (__bf16*)(ws + 0 * SZ);
  __bf16* LNk = (__bf16*)(ws + 1 * SZ);
  __bf16* LNv = (__bf16*)(ws + 2 * SZ);
  __bf16* Qh  = (__bf16*)(ws + 3 * SZ);
  __bf16* Kh  = (__bf16*)(ws + 4 * SZ);
  __bf16* Vh  = (__bf16*)(ws + 5 * SZ);
  __bf16* AO  = (__bf16*)(ws + 6 * SZ);

  mha_ln_bf16<<<dim3(MROWS, 3), 256, 0, stream>>>(q, k, v, lng, lnb, LNq, LNk, LNv);

  const dim3 gg(HID / TN, MROWS / TM);  // (8, 32)
  mha_gemm512<<<gg, 256, 0, stream>>>(LNq, Wq, bq, Qh, nullptr);
  mha_gemm512<<<gg, 256, 0, stream>>>(LNk, Wk, bk, Kh, nullptr);
  mha_gemm512<<<gg, 256, 0, stream>>>(LNv, Wv, bv, Vh, nullptr);

  mha_flash_attn<<<dim3(SEQ / 128, BAT * NH), 256, 0, stream>>>(Qh, Kh, Vh, AO);

  mha_gemm512<<<gg, 256, 0, stream>>>(AO, Wo, bo, nullptr, (float*)d_out);
}
